// ParallelTransformerBlock_84610855731420
// MI455X (gfx1250) — matrix operand order repaired and checked by reading, not compiled, not run on hardware
//
#include <hip/hip_runtime.h>

// ---------------------------------------------------------------------------
// CDNA5 (gfx1250) parallel transformer block, bf16 WMMA (16x16x32), wave32.
// Round 2: 32x64 wave tiles in both big GEMMs (2x FLOP/byte), and
// GLOBAL_LOAD_ASYNC_TO_LDS_B128 staging of the K chunk in attention.
// ---------------------------------------------------------------------------

typedef __bf16 bf16;
typedef __attribute__((ext_vector_type(8)))  __bf16 v8bf;
typedef __attribute__((ext_vector_type(16))) __bf16 v16bf;
typedef __attribute__((ext_vector_type(8)))  float  v8f;

#define NTOK        2048
#define DIM         2048
#define DH          128
#define NHEADS      16
#define ATTN_INNER  2048           // NHEADS * DH
#define FFD         8192           // FF_MULT * DIM
#define FUSED_N     18688          // ATTN_INNER + DH + DH + 2*FFD
#define QSCALE      0.08838834764831845f   // 1/sqrt(128)

#define WMMA_BF16(a, b, c) \
  __builtin_amdgcn_wmma_f32_16x16x32_bf16(false, (a), false, (b), (short)0, (c), false, false)

#if __has_builtin(__builtin_amdgcn_s_wait_asynccnt)
#define ASYNC_WAIT0() __builtin_amdgcn_s_wait_asynccnt(0)
#else
#define ASYNC_WAIT0() asm volatile("s_wait_asynccnt 0x0" ::: "memory")
#endif

// ---- A fragment (16x32 bf16, row-major source) ----------------------------
// Per ISA 7.12.2: lanes 0-15 row M=lane hold K = {kb+0..7, kb+16..23};
//                 lanes 16-31 row M=lane-16 hold K = {kb+8..15, kb+24..31}.
__device__ __forceinline__ v16bf frag_a(const bf16* rowp, int kbase, int lane) {
  int off = kbase + ((lane & 16) ? 8 : 0);
  v8bf lo = *(const v8bf*)(rowp + off);
  v8bf hi = *(const v8bf*)(rowp + off + 16);
  return __builtin_shufflevector(lo, hi, 0, 1, 2, 3, 4, 5, 6, 7,
                                 8, 9, 10, 11, 12, 13, 14, 15);
}

// ---- B fragment (32x16 bf16) from B^T stored [N][K] row-major -------------
// Per ISA: lanes 0-15 col N=lane hold K = {kb+0..7, kb+16..23}; lanes 16-31 hold {kb+8..15, kb+24..31}.
__device__ __forceinline__ v16bf frag_b(const bf16* colp, int kbase, int lane) {
  int off = kbase + ((lane & 16) ? 8 : 0);
  v8bf lo = *(const v8bf*)(colp + off);
  v8bf hi = *(const v8bf*)(colp + off + 16);
  return __builtin_shufflevector(lo, hi, 0, 1, 2, 3, 4, 5, 6, 7,
                                 8, 9, 10, 11, 12, 13, 14, 15);
}

// ---------------------------------------------------------------------------
// 1) LayerNorm -> bf16 activations
// ---------------------------------------------------------------------------
__global__ __launch_bounds__(256) void ln_kernel(const float* __restrict__ x,
                                                 const float* __restrict__ gamma,
                                                 bf16* __restrict__ xn) {
  int tok = blockIdx.x;
  const float* row = x + (size_t)tok * DIM;
  float s = 0.f, ss = 0.f;
  for (int i = threadIdx.x; i < DIM; i += 256) {
    float v = row[i];
    s += v;
    ss += v * v;
  }
  for (int o = 16; o > 0; o >>= 1) {
    s  += __shfl_down(s, o, 32);
    ss += __shfl_down(ss, o, 32);
  }
  __shared__ float red0[8], red1[8], stats[2];
  int wave = threadIdx.x >> 5;
  if ((threadIdx.x & 31) == 0) { red0[wave] = s; red1[wave] = ss; }
  __syncthreads();
  if (threadIdx.x == 0) {
    float a = 0.f, b = 0.f;
    for (int i = 0; i < 8; i++) { a += red0[i]; b += red1[i]; }
    float mean = a * (1.0f / DIM);
    float var  = b * (1.0f / DIM) - mean * mean;
    stats[0] = mean;
    stats[1] = rsqrtf(var + 1e-5f);
  }
  __syncthreads();
  float mean = stats[0], inv = stats[1];
  bf16* orow = xn + (size_t)tok * DIM;
  for (int i = threadIdx.x; i < DIM; i += 256)
    orow[i] = (bf16)((row[i] - mean) * inv * gamma[i]);
}

// ---------------------------------------------------------------------------
// 2) Transpose + f32 -> bf16 convert: out[c][r] = in[r][c]
// ---------------------------------------------------------------------------
__global__ __launch_bounds__(256) void transpose_convert_kernel(
    const float* __restrict__ in, bf16* __restrict__ out, int R, int C) {
  __shared__ float tile[32][33];
  int c0 = blockIdx.x * 32, r0 = blockIdx.y * 32;
  int tx = threadIdx.x & 31, ty = threadIdx.x >> 5;
  for (int i = ty; i < 32; i += 8)
    tile[i][tx] = in[(size_t)(r0 + i) * C + c0 + tx];
  __syncthreads();
  for (int i = ty; i < 32; i += 8)
    out[(size_t)(c0 + i) * R + r0 + tx] = (bf16)tile[tx][i];
}

// ---------------------------------------------------------------------------
// 3) GEMM: proj[M][N] = A[M][K] @ Bt[N][K]^T   (bf16 out)
//    block = 4 waves; each wave computes a 32x64 tile (8 WMMAs / k-step).
// ---------------------------------------------------------------------------
__global__ __launch_bounds__(128) void gemm_xw_kernel(const bf16* __restrict__ A,
                                                      const bf16* __restrict__ Bt,
                                                      bf16* __restrict__ C,
                                                      int N, int K) {
  int lane = threadIdx.x & 31;
  int wave = threadIdx.x >> 5;
  int m0 = blockIdx.y * 32;
  int n0 = (blockIdx.x * 4 + wave) * 64;
  const bf16* arow0 = A + (size_t)(m0 + (lane & 15)) * K;
  const bf16* arow1 = arow0 + (size_t)16 * K;
  const bf16* bcol  = Bt + (size_t)(n0 + (lane & 15)) * K;
  v8f acc[2][4];
#pragma unroll
  for (int i = 0; i < 2; i++)
#pragma unroll
    for (int t = 0; t < 4; t++) acc[i][t] = (v8f){};
  for (int k = 0; k < K; k += 32) {
    v16bf a0 = frag_a(arow0, k, lane);
    v16bf a1 = frag_a(arow1, k, lane);
    v16bf b0 = frag_b(bcol, k, lane);
    v16bf b1 = frag_b(bcol + (size_t)16 * K, k, lane);
    v16bf b2 = frag_b(bcol + (size_t)32 * K, k, lane);
    v16bf b3 = frag_b(bcol + (size_t)48 * K, k, lane);
    acc[0][0] = WMMA_BF16(a0, b0, acc[0][0]);
    acc[0][1] = WMMA_BF16(a0, b1, acc[0][1]);
    acc[0][2] = WMMA_BF16(a0, b2, acc[0][2]);
    acc[0][3] = WMMA_BF16(a0, b3, acc[0][3]);
    acc[1][0] = WMMA_BF16(a1, b0, acc[1][0]);
    acc[1][1] = WMMA_BF16(a1, b1, acc[1][1]);
    acc[1][2] = WMMA_BF16(a1, b2, acc[1][2]);
    acc[1][3] = WMMA_BF16(a1, b3, acc[1][3]);
  }
  int cm = (lane & 16) ? 8 : 0, cn = lane & 15;
#pragma unroll
  for (int i = 0; i < 2; i++) {
#pragma unroll
    for (int r = 0; r < 8; r++) {
      bf16* crow = C + (size_t)(m0 + i * 16 + cm + r) * N + n0 + cn;
      crow[0]  = (bf16)acc[i][0][r];
      crow[16] = (bf16)acc[i][1][r];
      crow[32] = (bf16)acc[i][2][r];
      crow[48] = (bf16)acc[i][3][r];
    }
  }
}

// ---------------------------------------------------------------------------
// 4) Pointwise: rotary(q)*scale, rotary(k), v -> v^T, silu(gate)*ff_x
// ---------------------------------------------------------------------------
__global__ __launch_bounds__(256) void pointwise_kernel(const bf16* __restrict__ proj,
                                                        bf16* __restrict__ q,
                                                        bf16* __restrict__ kk,
                                                        bf16* __restrict__ vt,
                                                        bf16* __restrict__ ffact) {
  int tok = blockIdx.x;
  const bf16* prow = proj + (size_t)tok * FUSED_N;
  float tf = (float)tok;
  for (int d = threadIdx.x; d < ATTN_INNER; d += 256) {
    int dd = d & 127;
    int j = dd & 63;
    float invf = __powf(10000.0f, -(float)(2 * j) * (1.0f / 128.0f));
    float sn, cs;
    __sincosf(tf * invf, &sn, &cs);
    float t  = (float)prow[d];
    float ot = (dd < 64) ? -(float)prow[d + 64] : (float)prow[d - 64];
    q[(size_t)tok * ATTN_INNER + d] = (bf16)((t * cs + ot * sn) * QSCALE);
  }
  for (int dd = threadIdx.x; dd < DH; dd += 256) {
    int j = dd & 63;
    float invf = __powf(10000.0f, -(float)(2 * j) * (1.0f / 128.0f));
    float sn, cs;
    __sincosf(tf * invf, &sn, &cs);
    float t  = (float)prow[ATTN_INNER + dd];
    float ot = (dd < 64) ? -(float)prow[ATTN_INNER + dd + 64]
                         :  (float)prow[ATTN_INNER + dd - 64];
    kk[(size_t)tok * DH + dd] = (bf16)(t * cs + ot * sn);
  }
  for (int dd = threadIdx.x; dd < DH; dd += 256)
    vt[(size_t)dd * NTOK + tok] = prow[ATTN_INNER + DH + dd];
  const bf16* pff = prow + ATTN_INNER + 2 * DH;
  for (int i = threadIdx.x; i < FFD; i += 256) {
    float fx = (float)pff[i];
    float g  = (float)pff[i + FFD];
    float act = g / (1.0f + __expf(-g));
    ffact[(size_t)tok * FFD + i] = (bf16)(act * fx);
  }
}

// ---------------------------------------------------------------------------
// 5) Flash attention: 1 wave per (16-query block, head), online softmax,
//    64-key chunks. K chunk staged into LDS via async global->LDS copies
//    (ASYNCcnt path); S tile staged in LDS; P re-read as WMMA A-fragments.
// ---------------------------------------------------------------------------
__global__ __launch_bounds__(32) void attn_kernel(const bf16* __restrict__ Q,
                                                  const bf16* __restrict__ Kk,
                                                  const bf16* __restrict__ Vt,
                                                  bf16* __restrict__ O) {
  int lane = threadIdx.x;
  int qb = blockIdx.x;   // 16-row query block
  int h  = blockIdx.y;   // head
  __shared__ bf16  Ksh[64][DH];   // 16 KB staged key chunk
  __shared__ float Sb[16][64];
  __shared__ bf16  Pb[16][64];
  __shared__ float rowtmp[16];

  float mrow = -3.0e38f, lrow = 0.0f;   // valid on lanes 0..15 (row = lane)
  v8f acc[8];
#pragma unroll
  for (int t = 0; t < 8; t++) acc[t] = (v8f){};

  const bf16* qrow = Q + (size_t)(qb * 16 + (lane & 15)) * ATTN_INNER + h * DH;
  unsigned lds_base = (unsigned)(uintptr_t)&Ksh[0][0];
  int kend = qb * 16 + 16;
  int nchunks = (kend + 63) / 64;

  for (int ch = 0; ch < nchunks; ch++) {
    int kc0 = ch * 64;
    // --- async stage 32 keys x 128 dh (8 KB) into LDS ---
    {
      const bf16* gsrc = Kk + (size_t)kc0 * DH;
#pragma unroll
      for (int it = 0; it < 16; it++) {
        unsigned lds = lds_base + (unsigned)(it * 512 + lane * 16);
        const bf16* g = gsrc + it * 256 + lane * 8;
        asm volatile("global_load_async_to_lds_b128 %0, %1, off"
                     :: "v"(lds), "v"(g)
                     : "memory");
      }
      ASYNC_WAIT0();
    }
    // --- S = q @ k^T over 64 keys (4 column tiles, K = 128), B from LDS ---
#pragma unroll
    for (int nt = 0; nt < 4; nt++) {
      const bf16* krow = &Ksh[nt * 16 + (lane & 15)][0];
      v8f s = {};
#pragma unroll
      for (int kkk = 0; kkk < DH; kkk += 32) {
        v16bf a = frag_a(qrow, kkk, lane);
        v16bf b = frag_b(krow, kkk, lane);
        s = WMMA_BF16(a, b, s);
      }
      int cm = (lane & 16) ? 8 : 0, cn = lane & 15;
#pragma unroll
      for (int r = 0; r < 8; r++) Sb[cm + r][nt * 16 + cn] = s[r];
    }
    __syncthreads();
    // --- online softmax row update (lanes 0..15 own one row each) ---
    if (lane < 16) {
      int qi = qb * 16 + lane;
      int kvalid = qi - kc0 + 1;          // causal: keys <= query
      if (kvalid > 64) kvalid = 64;       // kc0 <= qb*16 guarantees kvalid >= 1
      float cmax = -3.0e38f;
      for (int c = 0; c < kvalid; c++) cmax = fmaxf(cmax, Sb[lane][c]);
      float mnew  = fmaxf(mrow, cmax);
      float alpha = __expf(mrow - mnew);
      float rs = 0.f;
      for (int c = 0; c < 64; c++) {
        float p = (c < kvalid) ? __expf(Sb[lane][c] - mnew) : 0.0f;
        Pb[lane][c] = (bf16)p;
        rs += p;
      }
      lrow = lrow * alpha + rs;
      mrow = mnew;
      rowtmp[lane] = alpha;
    }
    __syncthreads();
    // --- rescale running O ---
    {
      int cm = (lane & 16) ? 8 : 0;
#pragma unroll
      for (int r = 0; r < 8; r++) {
        float a = rowtmp[cm + r];
#pragma unroll
        for (int t = 0; t < 8; t++) acc[t][r] *= a;
      }
    }
    // --- O += P(16x64) @ V(64x128) ---
    const bf16* prow = &Pb[lane & 15][0];
#pragma unroll
    for (int kkk = 0; kkk < 64; kkk += 32) {
      v16bf a = frag_a(prow, kkk, lane);
#pragma unroll
      for (int t = 0; t < 8; t++) {
        const bf16* vrow = Vt + (size_t)(t * 16 + (lane & 15)) * NTOK + kc0;
        v16bf b = frag_b(vrow, kkk, lane);
        acc[t] = WMMA_BF16(a, b, acc[t]);
      }
    }
    __syncthreads();
  }
  // --- normalize by row sums and store ---
  if (lane < 16) rowtmp[lane] = 1.0f / lrow;
  __syncthreads();
  int cm = (lane & 16) ? 8 : 0, cn = lane & 15;
#pragma unroll
  for (int r = 0; r < 8; r++) {
    float inv = rowtmp[cm + r];
    int tok = qb * 16 + cm + r;
    bf16* orow = O + (size_t)tok * ATTN_INNER + h * DH + cn;
#pragma unroll
    for (int t = 0; t < 8; t++) orow[t * 16] = (bf16)(acc[t][r] * inv);
  }
}

// ---------------------------------------------------------------------------
// 6) Fused output GEMM (32x64 wave tiles):
//    out = O[2048x2048] @ Wao_t^T  +  FF[2048x8192] @ Wff_t^T   (f32 out)
// ---------------------------------------------------------------------------
__global__ __launch_bounds__(128) void gemm_out_kernel(const bf16* __restrict__ A1,
                                                       const bf16* __restrict__ B1t,
                                                       const bf16* __restrict__ A2,
                                                       const bf16* __restrict__ B2t,
                                                       float* __restrict__ C) {
  int lane = threadIdx.x & 31;
  int wave = threadIdx.x >> 5;
  int m0 = blockIdx.y * 32;
  int n0 = (blockIdx.x * 4 + wave) * 64;
  v8f acc[2][4];
#pragma unroll
  for (int i = 0; i < 2; i++)
#pragma unroll
    for (int t = 0; t < 4; t++) acc[i][t] = (v8f){};
  {
    const bf16* ar0 = A1 + (size_t)(m0 + (lane & 15)) * ATTN_INNER;
    const bf16* ar1 = ar0 + (size_t)16 * ATTN_INNER;
    const bf16* br  = B1t + (size_t)(n0 + (lane & 15)) * ATTN_INNER;
    for (int k = 0; k < ATTN_INNER; k += 32) {
      v16bf a0 = frag_a(ar0, k, lane);
      v16bf a1 = frag_a(ar1, k, lane);
#pragma unroll
      for (int t = 0; t < 4; t++) {
        v16bf b = frag_b(br + (size_t)(t * 16) * ATTN_INNER, k, lane);
        acc[0][t] = WMMA_BF16(a0, b, acc[0][t]);
        acc[1][t] = WMMA_BF16(a1, b, acc[1][t]);
      }
    }
  }
  {
    const bf16* ar0 = A2 + (size_t)(m0 + (lane & 15)) * FFD;
    const bf16* ar1 = ar0 + (size_t)16 * FFD;
    const bf16* br  = B2t + (size_t)(n0 + (lane & 15)) * FFD;
    for (int k = 0; k < FFD; k += 32) {
      v16bf a0 = frag_a(ar0, k, lane);
      v16bf a1 = frag_a(ar1, k, lane);
#pragma unroll
      for (int t = 0; t < 4; t++) {
        v16bf b = frag_b(br + (size_t)(t * 16) * FFD, k, lane);
        acc[0][t] = WMMA_BF16(a0, b, acc[0][t]);
        acc[1][t] = WMMA_BF16(a1, b, acc[1][t]);
      }
    }
  }
  int cm = (lane & 16) ? 8 : 0, cn = lane & 15;
#pragma unroll
  for (int i = 0; i < 2; i++) {
#pragma unroll
    for (int r = 0; r < 8; r++) {
      float* crow = C + (size_t)(m0 + i * 16 + cm + r) * DIM + n0 + cn;
      crow[0]  = acc[i][0][r];
      crow[16] = acc[i][1][r];
      crow[32] = acc[i][2][r];
      crow[48] = acc[i][3][r];
    }
  }
}

// ---------------------------------------------------------------------------
// Host-side orchestration
// ---------------------------------------------------------------------------
extern "C" void kernel_launch(void* const* d_in, const int* in_sizes, int n_in,
                              void* d_out, int out_size, void* d_ws, size_t ws_size,
                              hipStream_t stream) {
  (void)in_sizes; (void)n_in; (void)out_size; (void)ws_size;
  const float* x          = (const float*)d_in[0];
  const float* gamma      = (const float*)d_in[1];
  const float* w_fused    = (const float*)d_in[2];
  const float* w_attn_out = (const float*)d_in[3];
  const float* w_ff_out   = (const float*)d_in[4];
  float* out = (float*)d_out;

  char* base = (char*)d_ws;
  size_t off = 0;
  auto alloc = [&](size_t bytes) -> char* {
    char* p = base + off;
    off += (bytes + 255) & ~(size_t)255;
    return p;
  };
  bf16* xn       = (bf16*)alloc((size_t)NTOK * DIM * 2);
  bf16* wfused_t = (bf16*)alloc((size_t)FUSED_N * DIM * 2);
  bf16* wao_t    = (bf16*)alloc((size_t)DIM * ATTN_INNER * 2);
  bf16* wff_t    = (bf16*)alloc((size_t)DIM * FFD * 2);
  bf16* proj     = (bf16*)alloc((size_t)NTOK * FUSED_N * 2);
  bf16* q        = (bf16*)alloc((size_t)NTOK * ATTN_INNER * 2);
  bf16* kk       = (bf16*)alloc((size_t)NTOK * DH * 2);
  bf16* vt       = (bf16*)alloc((size_t)DH * NTOK * 2);
  bf16* ffact    = (bf16*)alloc((size_t)NTOK * FFD * 2);
  bf16* attno    = (bf16*)alloc((size_t)NTOK * ATTN_INNER * 2);

  ln_kernel<<<NTOK, 256, 0, stream>>>(x, gamma, xn);
  transpose_convert_kernel<<<dim3(FUSED_N / 32, DIM / 32), 256, 0, stream>>>(
      w_fused, wfused_t, DIM, FUSED_N);
  transpose_convert_kernel<<<dim3(DIM / 32, ATTN_INNER / 32), 256, 0, stream>>>(
      w_attn_out, wao_t, ATTN_INNER, DIM);
  transpose_convert_kernel<<<dim3(DIM / 32, FFD / 32), 256, 0, stream>>>(
      w_ff_out, wff_t, FFD, DIM);
  gemm_xw_kernel<<<dim3(FUSED_N / 256, NTOK / 32), 128, 0, stream>>>(
      xn, wfused_t, proj, FUSED_N, DIM);
  pointwise_kernel<<<NTOK, 256, 0, stream>>>(proj, q, kk, vt, ffact);
  attn_kernel<<<dim3(NTOK / 16, NHEADS), 32, 0, stream>>>(q, kk, vt, attno);
  gemm_out_kernel<<<dim3(DIM / 256, NTOK / 32), 128, 0, stream>>>(
      attno, wao_t, ffact, wff_t, out);
}
